// AttentionBlock_77816217469131
// MI455X (gfx1250) — matrix operand order repaired and checked by reading, not compiled, not run on hardware
//
#include <hip/hip_runtime.h>

typedef __attribute__((ext_vector_type(16))) _Float16 v16h;
typedef __attribute__((ext_vector_type(8)))  _Float16 v8h;
typedef __attribute__((ext_vector_type(8)))  float    v8f;

#define NB_  8
#define C_   512
#define H_   8
#define D_   64
#define HW_  1024
#define CPG_ 16
#define EPS_ 1e-5f
#define QK_SCALE 0.35355339059327373f   // 64^-0.25

static __device__ __forceinline__ v16h cat8(v8h lo, v8h hi) {
    v16h r;
#pragma unroll
    for (int i = 0; i < 8; ++i) { r[i] = lo[i]; r[8 + i] = hi[i]; }
    return r;
}

static __device__ __forceinline__ v8f wmma16(v16h a, v16h b, v8f c) {
    return __builtin_amdgcn_wmma_f32_16x16x32_f16(false, a, false, b, (short)0, c, false, false);
}

// async 16B/lane global -> LDS copy (ASYNCcnt tracked)
static __device__ __forceinline__ void async_g2l_b128(unsigned lds_off, const void* gaddr) {
    asm volatile("global_load_async_to_lds_b128 %0, %1, off"
                 :: "v"(lds_off), "v"((unsigned long long)(uintptr_t)gaddr)
                 : "memory");
}
template <int N>
static __device__ __forceinline__ void async_wait_le() {
    asm volatile("s_wait_asynccnt %0" :: "n"(N) : "memory");
}
static __device__ __forceinline__ unsigned lds_off_of(const void* p) {
    return (unsigned)(uintptr_t)p;   // low 32 bits of flat shared addr == LDS offset
}

// ---------------------------------------------------------------- weights -> f16
__global__ void cvt_w_kernel(const float* __restrict__ wq, const float* __restrict__ wo,
                             _Float16* __restrict__ wqh, _Float16* __restrict__ woh) {
    int i = blockIdx.x * 256 + threadIdx.x;
    if (i < 3 * C_ * C_) wqh[i] = (_Float16)wq[i];
    if (i < C_ * C_)     woh[i] = (_Float16)wo[i];
}

// ---------------------------------------------------------------- GroupNorm 1
__global__ void gn1_kernel(const float* __restrict__ x, const float* __restrict__ scale,
                           const float* __restrict__ offset,
                           float* __restrict__ xn, _Float16* __restrict__ xnh) {
    int n = blockIdx.x >> 5, g = blockIdx.x & 31, t = threadIdx.x;
    const float* xb = x + ((size_t)n * HW_) * C_ + g * CPG_;
    float s = 0.f, ss = 0.f;
    for (int p = t; p < HW_; p += 256) {
        const float4* rp = (const float4*)(xb + (size_t)p * C_);
#pragma unroll
        for (int j = 0; j < 4; ++j) {
            float4 v = rp[j];
            s  += v.x + v.y + v.z + v.w;
            ss += v.x * v.x + v.y * v.y + v.z * v.z + v.w * v.w;
        }
    }
    __shared__ float rs[256], rss[256];
    rs[t] = s; rss[t] = ss; __syncthreads();
    for (int st = 128; st > 0; st >>= 1) {
        if (t < st) { rs[t] += rs[t + st]; rss[t] += rss[t + st]; }
        __syncthreads();
    }
    float mean = rs[0] * (1.f / 16384.f);
    float var  = rss[0] * (1.f / 16384.f) - mean * mean;
    float rstd = rsqrtf(var + EPS_);
    float sc[CPG_], of[CPG_];
#pragma unroll
    for (int j = 0; j < CPG_; ++j) {
        float scl = scale[g * CPG_ + j];
        sc[j] = scl * rstd;
        of[j] = offset[g * CPG_ + j] - mean * scl * rstd;
    }
    float*    xnb = xn  + ((size_t)n * HW_) * C_ + g * CPG_;
    _Float16* xhb = xnh + ((size_t)n * HW_) * C_ + g * CPG_;
    for (int p = t; p < HW_; p += 256) {
        const float* rp = xb + (size_t)p * C_;
        float* wp = xnb + (size_t)p * C_;
        _Float16* hp = xhb + (size_t)p * C_;
#pragma unroll
        for (int j = 0; j < CPG_; ++j) {
            float v = rp[j] * sc[j] + of[j];
            wp[j] = v; hp[j] = (_Float16)v;
        }
    }
}

// ---------------------------------------------------------------- QKV GEMM (WMMA)
// C[8192 x 1536] = A[8192 x 512] * B^T, B = w_qkv [1536 x 512] row-major ([N,K])
// block: 128 thr = 4 waves; block tile 128x64; wave tile 32x64; double-buffered async LDS
__global__ __launch_bounds__(128) void qkv_gemm_kernel(
        const _Float16* __restrict__ A, const _Float16* __restrict__ B,
        const float* __restrict__ bias,
        _Float16* __restrict__ qb, _Float16* __restrict__ kb, _Float16* __restrict__ vb) {
    __shared__ __align__(16) _Float16 As[2][128][48];
    __shared__ __align__(16) _Float16 Bs[2][64][48];
    const int w = threadIdx.x >> 5, lane = threadIdx.x & 31, t = threadIdx.x;
    const int m0 = blockIdx.y * 128, n0 = blockIdx.x * 64;
    v8f acc[2][4] = {};

    auto issue_chunk = [&](int kk, int buf) {
#pragma unroll
        for (int it = 0; it < 4; ++it) {                  // A: 128 rows x 4 chunks
            int idx = t + it * 128;
            int row = idx >> 2, c8 = (idx & 3) * 8;
            async_g2l_b128(lds_off_of(&As[buf][row][c8]), A + (size_t)(m0 + row) * C_ + kk + c8);
        }
#pragma unroll
        for (int it = 0; it < 2; ++it) {                  // B: 64 rows x 4 chunks
            int idx = t + it * 128;
            int row = idx >> 2, c8 = (idx & 3) * 8;
            async_g2l_b128(lds_off_of(&Bs[buf][row][c8]), B + (size_t)(n0 + row) * C_ + kk + c8);
        }
    };

    issue_chunk(0, 0);
    for (int i = 0; i < 16; ++i) {
        int cur = i & 1;
        if (i < 15) { issue_chunk((i + 1) * 32, cur ^ 1); async_wait_le<6>(); }
        else        { async_wait_le<0>(); }
        __syncthreads();
        v16h af[2];
#pragma unroll
        for (int mi = 0; mi < 2; ++mi) {
            int ar = w * 32 + mi * 16 + (lane & 15);
            af[mi] = cat8(*(const v8h*)(&As[cur][ar][(lane >> 4) * 8]),
                          *(const v8h*)(&As[cur][ar][(lane >> 4) * 8 + 16]));
        }
#pragma unroll
        for (int f = 0; f < 4; ++f) {
            int br = f * 16 + (lane & 15);
            v16h bf = cat8(*(const v8h*)(&Bs[cur][br][(lane >> 4) * 8]),
                           *(const v8h*)(&Bs[cur][br][(lane >> 4) * 8 + 16]));
#pragma unroll
            for (int mi = 0; mi < 2; ++mi)
                acc[mi][f] = wmma16(af[mi], bf, acc[mi][f]);
        }
        __syncthreads();
    }
#pragma unroll
    for (int f = 0; f < 4; ++f) {
        int o = n0 + f * 16 + (lane & 15);
        float bv = bias[o];
        int sel = o >> 9, oo = o & 511;
        int head = oo >> 6, dim = oo & 63;
#pragma unroll
        for (int mi = 0; mi < 2; ++mi)
#pragma unroll
            for (int r = 0; r < 8; ++r) {
                int m = m0 + w * 32 + mi * 16 + r + (lane >> 4) * 8;
                int nn = m >> 10, p = m & 1023;
                float v = acc[mi][f][r] + bv;
                if (sel == 0)
                    qb[(((size_t)nn * H_ + head) * HW_ + p) * D_ + dim] = (_Float16)(v * QK_SCALE);
                else if (sel == 1)
                    kb[(((size_t)nn * H_ + head) * HW_ + p) * D_ + dim] = (_Float16)(v * QK_SCALE);
                else
                    vb[(((size_t)nn * H_ + head) * D_ + dim) * HW_ + p] = (_Float16)v;
            }
    }
}

// ---------------------------------------------------------------- flash attention (WMMA)
__global__ __launch_bounds__(128) void attn_kernel(
        const _Float16* __restrict__ q, const _Float16* __restrict__ k,
        const _Float16* __restrict__ vT, _Float16* __restrict__ ychan) {
    __shared__ __align__(16) _Float16 Ks[2][64][72];
    __shared__ __align__(16) _Float16 Vs[2][64][72];
    __shared__ __align__(16) _Float16 Ps[4][16][72];
    const int w = threadIdx.x >> 5, lane = threadIdx.x & 31, t = threadIdx.x;
    const int nh = blockIdx.x >> 4;
    const int t0 = (blockIdx.x & 15) * 64;
    const _Float16* qbase = q  + (size_t)nh * HW_ * D_;
    const _Float16* kbase = k  + (size_t)nh * HW_ * D_;
    const _Float16* vbase = vT + (size_t)nh * D_ * HW_;
    int qrow = t0 + w * 16 + (lane & 15);
    v16h qf[2];
#pragma unroll
    for (int kc = 0; kc < 2; ++kc) {
        const _Float16* rp = qbase + (size_t)qrow * D_ + kc * 32 + (lane >> 4) * 8;
        qf[kc] = cat8(*(const v8h*)rp, *(const v8h*)(rp + 16));
    }
    float mrow[8], lrow[8];
    v8f yacc[4] = {};
#pragma unroll
    for (int r = 0; r < 8; ++r) { mrow[r] = -3.0e38f; lrow[r] = 0.f; }

    auto issue_chunk = [&](int s0, int buf) {
#pragma unroll
        for (int it = 0; it < 4; ++it) {
            int idx = t + it * 128;            // 0..511
            int row = idx >> 3, c8 = (idx & 7) * 8;
            async_g2l_b128(lds_off_of(&Ks[buf][row][c8]), kbase + (size_t)(s0 + row) * D_ + c8);
            async_g2l_b128(lds_off_of(&Vs[buf][row][c8]), vbase + (size_t)row * HW_ + s0 + c8);
        }
    };

    issue_chunk(0, 0);
    for (int i = 0; i < 16; ++i) {
        int cur = i & 1;
        if (i < 15) { issue_chunk((i + 1) * 64, cur ^ 1); async_wait_le<8>(); }
        else        { async_wait_le<0>(); }
        __syncthreads();
        // S tile 16t x 64s
        v8f sfr[4];
#pragma unroll
        for (int f = 0; f < 4; ++f) {
            v8f c = {};
#pragma unroll
            for (int kc = 0; kc < 2; ++kc) {
                const _Float16* rp = &Ks[cur][f * 16 + (lane & 15)][kc * 32 + (lane >> 4) * 8];
                c = wmma16(qf[kc], cat8(*(const v8h*)rp, *(const v8h*)(rp + 16)), c);
            }
            sfr[f] = c;
        }
        // online softmax
        float alpha[8];
#pragma unroll
        for (int r = 0; r < 8; ++r) {
            float mx = fmaxf(fmaxf(sfr[0][r], sfr[1][r]), fmaxf(sfr[2][r], sfr[3][r]));
#pragma unroll
            for (int msk = 8; msk > 0; msk >>= 1) mx = fmaxf(mx, __shfl_xor(mx, msk, 32));
            float mnew = fmaxf(mrow[r], mx);
            alpha[r] = __expf(mrow[r] - mnew);
            mrow[r] = mnew;
        }
#pragma unroll
        for (int f = 0; f < 4; ++f)
#pragma unroll
            for (int r = 0; r < 8; ++r) sfr[f][r] = __expf(sfr[f][r] - mrow[r]);
#pragma unroll
        for (int r = 0; r < 8; ++r) {
            float s = sfr[0][r] + sfr[1][r] + sfr[2][r] + sfr[3][r];
#pragma unroll
            for (int msk = 8; msk > 0; msk >>= 1) s += __shfl_xor(s, msk, 32);
            lrow[r] = lrow[r] * alpha[r] + s;
        }
#pragma unroll
        for (int f = 0; f < 4; ++f)
#pragma unroll
            for (int r = 0; r < 8; ++r) yacc[f][r] *= alpha[r];
        // stage P (D-layout -> A-layout via LDS, wave-private region)
#pragma unroll
        for (int f = 0; f < 4; ++f) {
            int col = f * 16 + (lane & 15);
#pragma unroll
            for (int r = 0; r < 8; ++r)
                Ps[w][r + (lane >> 4) * 8][col] = (_Float16)sfr[f][r];
        }
        v16h pf[2];
#pragma unroll
        for (int kc = 0; kc < 2; ++kc) {
            const _Float16* rp = &Ps[w][lane & 15][kc * 32 + (lane >> 4) * 8];
            pf[kc] = cat8(*(const v8h*)rp, *(const v8h*)(rp + 16));
        }
#pragma unroll
        for (int nd = 0; nd < 4; ++nd)
#pragma unroll
            for (int kc = 0; kc < 2; ++kc) {
                const _Float16* rp = &Vs[cur][nd * 16 + (lane & 15)][kc * 32 + (lane >> 4) * 8];
                yacc[nd] = wmma16(pf[kc], cat8(*(const v8h*)rp, *(const v8h*)(rp + 16)), yacc[nd]);
            }
        __syncthreads();
    }
    int nn = nh >> 3, head = nh & 7;
#pragma unroll
    for (int nd = 0; nd < 4; ++nd) {
        int dim = nd * 16 + (lane & 15);
#pragma unroll
        for (int r = 0; r < 8; ++r) {
            int p = t0 + w * 16 + r + (lane >> 4) * 8;
            float v = yacc[nd][r] / lrow[r];
            ychan[(((size_t)nn << 10) + p) * C_ + head * D_ + dim] = (_Float16)v;
        }
    }
}

// ---------------------------------------------------------------- out-proj GEMM (WMMA)
__global__ __launch_bounds__(128) void out_gemm_kernel(
        const _Float16* __restrict__ A, const _Float16* __restrict__ B,
        const float* __restrict__ bias, float* __restrict__ Co) {
    __shared__ __align__(16) _Float16 As[2][128][48];
    __shared__ __align__(16) _Float16 Bs[2][64][48];
    const int w = threadIdx.x >> 5, lane = threadIdx.x & 31, t = threadIdx.x;
    const int m0 = blockIdx.y * 128, n0 = blockIdx.x * 64;
    v8f acc[2][4] = {};

    auto issue_chunk = [&](int kk, int buf) {
#pragma unroll
        for (int it = 0; it < 4; ++it) {
            int idx = t + it * 128;
            int row = idx >> 2, c8 = (idx & 3) * 8;
            async_g2l_b128(lds_off_of(&As[buf][row][c8]), A + (size_t)(m0 + row) * C_ + kk + c8);
        }
#pragma unroll
        for (int it = 0; it < 2; ++it) {
            int idx = t + it * 128;
            int row = idx >> 2, c8 = (idx & 3) * 8;
            async_g2l_b128(lds_off_of(&Bs[buf][row][c8]), B + (size_t)(n0 + row) * C_ + kk + c8);
        }
    };

    issue_chunk(0, 0);
    for (int i = 0; i < 16; ++i) {
        int cur = i & 1;
        if (i < 15) { issue_chunk((i + 1) * 32, cur ^ 1); async_wait_le<6>(); }
        else        { async_wait_le<0>(); }
        __syncthreads();
        v16h af[2];
#pragma unroll
        for (int mi = 0; mi < 2; ++mi) {
            int ar = w * 32 + mi * 16 + (lane & 15);
            af[mi] = cat8(*(const v8h*)(&As[cur][ar][(lane >> 4) * 8]),
                          *(const v8h*)(&As[cur][ar][(lane >> 4) * 8 + 16]));
        }
#pragma unroll
        for (int f = 0; f < 4; ++f) {
            int br = f * 16 + (lane & 15);
            v16h bf = cat8(*(const v8h*)(&Bs[cur][br][(lane >> 4) * 8]),
                           *(const v8h*)(&Bs[cur][br][(lane >> 4) * 8 + 16]));
#pragma unroll
            for (int mi = 0; mi < 2; ++mi)
                acc[mi][f] = wmma16(af[mi], bf, acc[mi][f]);
        }
        __syncthreads();
    }
#pragma unroll
    for (int f = 0; f < 4; ++f) {
        int o = n0 + f * 16 + (lane & 15);
        float bv = bias[o];
#pragma unroll
        for (int mi = 0; mi < 2; ++mi)
#pragma unroll
            for (int r = 0; r < 8; ++r) {
                int m = m0 + w * 32 + mi * 16 + r + (lane >> 4) * 8;
                Co[(size_t)m * C_ + o] = acc[mi][f][r] + bv;
            }
    }
}

// ---------------------------------------------------------------- GroupNorm 2 + residual
__global__ void gn2_kernel(const float* __restrict__ og, const float* __restrict__ xn,
                           const float* __restrict__ scale, const float* __restrict__ offset,
                           float* __restrict__ out) {
    int n = blockIdx.x >> 5, g = blockIdx.x & 31, t = threadIdx.x;
    const float* ob = og + ((size_t)n * HW_) * C_ + g * CPG_;
    float s = 0.f, ss = 0.f;
    for (int p = t; p < HW_; p += 256) {
        const float4* rp = (const float4*)(ob + (size_t)p * C_);
#pragma unroll
        for (int j = 0; j < 4; ++j) {
            float4 v = rp[j];
            s  += v.x + v.y + v.z + v.w;
            ss += v.x * v.x + v.y * v.y + v.z * v.z + v.w * v.w;
        }
    }
    __shared__ float rs[256], rss[256];
    rs[t] = s; rss[t] = ss; __syncthreads();
    for (int st = 128; st > 0; st >>= 1) {
        if (t < st) { rs[t] += rs[t + st]; rss[t] += rss[t + st]; }
        __syncthreads();
    }
    float mean = rs[0] * (1.f / 16384.f);
    float var  = rss[0] * (1.f / 16384.f) - mean * mean;
    float rstd = rsqrtf(var + EPS_);
    float sc[CPG_], of[CPG_];
#pragma unroll
    for (int j = 0; j < CPG_; ++j) {
        float scl = scale[g * CPG_ + j];
        sc[j] = scl * rstd;
        of[j] = offset[g * CPG_ + j] - mean * scl * rstd;
    }
    const float* xb = xn + ((size_t)n * HW_) * C_ + g * CPG_;
    float* wb = out + ((size_t)n * HW_) * C_ + g * CPG_;
    for (int p = t; p < HW_; p += 256) {
        const float* rp = ob + (size_t)p * C_;
        const float* xp = xb + (size_t)p * C_;
        float* wp = wb + (size_t)p * C_;
#pragma unroll
        for (int j = 0; j < CPG_; ++j)
            wp[j] = xp[j] + rp[j] * sc[j] + of[j];
    }
}

// ---------------------------------------------------------------- launch
extern "C" void kernel_launch(void* const* d_in, const int* in_sizes, int n_in,
                              void* d_out, int out_size, void* d_ws, size_t ws_size,
                              hipStream_t stream) {
    const float* x     = (const float*)d_in[0];
    const float* gn1_s = (const float*)d_in[1];
    const float* gn1_o = (const float*)d_in[2];
    const float* w_qkv = (const float*)d_in[3];
    const float* b_qkv = (const float*)d_in[4];
    const float* w_out = (const float*)d_in[5];
    const float* b_out = (const float*)d_in[6];
    const float* gn2_s = (const float*)d_in[7];
    const float* gn2_o = (const float*)d_in[8];
    float* out = (float*)d_out;

    char* ws = (char*)d_ws;
    size_t off = 0;
    auto alloc = [&](size_t bytes) {
        off = (off + 255) & ~(size_t)255;
        void* p = ws + off; off += bytes; return p;
    };
    const size_t NPC = (size_t)NB_ * HW_ * C_;        // 8*1024*512
    float*    xn  = (float*)alloc(NPC * 4);
    float*    og  = (float*)alloc(NPC * 4);
    _Float16* xnh = (_Float16*)alloc(NPC * 2);
    _Float16* ych = (_Float16*)alloc(NPC * 2);
    _Float16* qb  = (_Float16*)alloc(NPC * 2);
    _Float16* kb  = (_Float16*)alloc(NPC * 2);
    _Float16* vb  = (_Float16*)alloc(NPC * 2);
    _Float16* wqh = (_Float16*)alloc((size_t)3 * C_ * C_ * 2);
    _Float16* woh = (_Float16*)alloc((size_t)C_ * C_ * 2);

    cvt_w_kernel<<<dim3(3072), dim3(256), 0, stream>>>(w_qkv, w_out, wqh, woh);
    gn1_kernel<<<dim3(256), dim3(256), 0, stream>>>(x, gn1_s, gn1_o, xn, xnh);
    qkv_gemm_kernel<<<dim3(24, 64), dim3(128), 0, stream>>>(xnh, wqh, b_qkv, qb, kb, vb);
    attn_kernel<<<dim3(1024), dim3(128), 0, stream>>>(qb, kb, vb, ych);
    out_gemm_kernel<<<dim3(8, 64), dim3(128), 0, stream>>>(ych, woh, b_out, og);
    gn2_kernel<<<dim3(256), dim3(256), 0, stream>>>(og, xn, gn2_s, gn2_o, out);
}
